// GPTLanguageModel_15582141350367
// MI455X (gfx1250) — matrix operand order repaired and checked by reading, not compiled, not run on hardware
//
#include <hip/hip_runtime.h>
#include <hip/hip_bf16.h>
#include <math.h>

// ---------------- model dims (from reference) ----------------
#define VSZ   32000
#define CDIM  1024
#define TSEQ  1024
#define NH    16
#define NL    4
#define NB    4
#define HDIM  64          // CDIM / NH
#define FFD   4096        // 4 * CDIM
#define MROWS (NB*TSEQ)   // 4096 token rows
#define LN_EPS 1e-5f

typedef __attribute__((ext_vector_type(8)))  __bf16 v8bf;
typedef __attribute__((ext_vector_type(16))) __bf16 v16bf;
typedef __attribute__((ext_vector_type(8)))  float  v8f;
typedef __attribute__((ext_vector_type(4)))  int    v4i;

#ifndef __has_builtin
#define __has_builtin(x) 0
#endif

// gfx1250 async global->LDS copies (guarded: falls back to sync copies)
#if __has_builtin(__builtin_amdgcn_global_load_async_to_lds_b128) && \
    __has_builtin(__builtin_amdgcn_s_wait_asynccnt)
#define USE_ASYNC 1
#else
#define USE_ASYNC 0
#endif

#if USE_ASYNC
// builtin signature (from compiler diagnostic): v4i* in AS1, v4i* in AS3, imm, imm
typedef __attribute__((address_space(1))) v4i v4i_as1;
typedef __attribute__((address_space(3))) v4i v4i_as3;
static __device__ __forceinline__ v4i_as1* to_gbl(const void* p) {
    return (v4i_as1*)(uintptr_t)p;
}
static __device__ __forceinline__ v4i_as3* to_lds(void* p) {
    return (v4i_as3*)(uint32_t)(uintptr_t)p;   // low 32 bits = LDS offset
}
#endif

// =====================================================================
// 1) token embedding + sinusoidal position
// =====================================================================
__global__ __launch_bounds__(256) void embed_kernel(
    const int* __restrict__ idx, const float* __restrict__ emb,
    const float* __restrict__ pos, float* __restrict__ x)
{
    int m   = blockIdx.x;
    int tid = threadIdx.x;
    int tok = idx[m];
    int t   = m % TSEQ;
    const float* e = emb + (size_t)tok * CDIM;
    const float* p = pos + (size_t)t   * CDIM;
    float* xr = x + (size_t)m * CDIM;
    for (int c = tid; c < CDIM; c += 256) xr[c] = e[c] + p[c];
}

// =====================================================================
// 2) LayerNorm (f32 in) -> bf16 out (GEMM A operand)
// =====================================================================
__global__ __launch_bounds__(256) void ln_bf16_kernel(
    const float* __restrict__ x, const float* __restrict__ g,
    const float* __restrict__ b, __bf16* __restrict__ out)
{
    int row = blockIdx.x, tid = threadIdx.x;
    const float* xr = x + (size_t)row * CDIM;
    float s = 0.f, s2 = 0.f;
    for (int c = tid; c < CDIM; c += 256) { float v = xr[c]; s += v; s2 += v * v; }
    __shared__ float rs[256], rq[256];
    rs[tid] = s; rq[tid] = s2; __syncthreads();
    for (int st = 128; st > 0; st >>= 1) {
        if (tid < st) { rs[tid] += rs[tid + st]; rq[tid] += rq[tid + st]; }
        __syncthreads();
    }
    float mean = rs[0] * (1.0f / CDIM);
    float var  = rq[0] * (1.0f / CDIM) - mean * mean;
    float inv  = rsqrtf(var + LN_EPS);
    __bf16* o = out + (size_t)row * CDIM;
    for (int c = tid; c < CDIM; c += 256)
        o[c] = (__bf16)((xr[c] - mean) * inv * g[c] + b[c]);
}

// =====================================================================
// 3) f32 weight [K,N] -> bf16 transposed weight Wt[N,K] (tiled via LDS)
// =====================================================================
__global__ __launch_bounds__(256) void cvt_transpose_kernel(
    const float* __restrict__ W, __bf16* __restrict__ Wt, int Kdim, int Ndim)
{
    __shared__ float tile[32][33];
    const int n0 = blockIdx.x * 32, k0 = blockIdx.y * 32;
    const int ln = threadIdx.x & 31, lr = threadIdx.x >> 5;   // lr: 0..7
#pragma unroll
    for (int r = lr; r < 32; r += 8)
        tile[r][ln] = W[(size_t)(k0 + r) * Ndim + n0 + ln];
    __syncthreads();
#pragma unroll
    for (int r = lr; r < 32; r += 8)
        Wt[(size_t)(n0 + r) * Kdim + k0 + ln] = (__bf16)tile[ln][r];
}

// =====================================================================
// 3b) per-head V transpose: vb[(b*T+t)*C + h*HD + d] -> vt[bh][d][t]
// =====================================================================
__global__ __launch_bounds__(256) void vtrans_kernel(
    const __bf16* __restrict__ vb, __bf16* __restrict__ vt)
{
    __shared__ __bf16 tile[32][33];
    const int t0 = blockIdx.x * 32;
    const int z  = blockIdx.y;              // bh*2 + dtile
    const int bh = z >> 1, dt = z & 1;
    const int b  = bh >> 4, h = bh & 15;
    const int d0 = dt * 32;
    const int ln = threadIdx.x & 31, lr = threadIdx.x >> 5;
    const size_t inBase = (size_t)b * TSEQ * CDIM + (size_t)h * HDIM;
#pragma unroll
    for (int r = lr; r < 32; r += 8)
        tile[r][ln] = vb[inBase + (size_t)(t0 + r) * CDIM + d0 + ln];
    __syncthreads();
    const size_t outBase = (size_t)bh * HDIM * TSEQ;
#pragma unroll
    for (int r = lr; r < 32; r += 8)
        vt[outBase + (size_t)(d0 + r) * TSEQ + t0 + ln] = tile[ln][r];
}

// =====================================================================
// 4) WMMA bf16 GEMM:  out = epi( A[M,K] @ Wt[N,K]^T + bias ) + residual
//    256 thr = 8 waves (2 along M x 4 along N); block tile 128x128;
//    double-buffered LDS staging (async-to-LDS when available).
// =====================================================================
__global__ __launch_bounds__(256) void gemm_bf16_wmma(
    const __bf16* __restrict__ A, const __bf16* __restrict__ Wt,
    const float* __restrict__ bias, const float* __restrict__ residual,
    float* __restrict__ outF, __bf16* __restrict__ outB,
    int Mdim, int Ndim, int Kdim, int doGelu)
{
    __shared__ __bf16 Alds[2][128 * 32];   // [buf][row 0..127][k 0..31]
    __shared__ __bf16 Blds[2][128 * 32];   // [buf][n   0..127][k 0..31]

    const int tid   = threadIdx.x;
    const int lane  = tid & 31;
    const int wave  = tid >> 5;
    const int mw    = wave & 1;            // 2 waves along M (64 rows each)
    const int nw    = wave >> 1;           // 4 waves along N (32 cols each)
    const int g     = lane >> 4;           // K-half selector
    const int ln    = lane & 15;
    const int nBase = blockIdx.x * 128;
    const int mBase = blockIdx.y * 128;

    const int srow  = tid >> 1, shalf = tid & 1;
    const __bf16* aSrc = A  + (size_t)(mBase + srow) * Kdim + shalf * 16;
    const __bf16* bSrc = Wt + (size_t)(nBase + srow) * Kdim + shalf * 16;
    const int     dOff = srow * 32 + shalf * 16;

    v8f acc[4][2] = {};

    auto stage = [&](int buf, int k0) {
        const __bf16* as = aSrc + k0;
        const __bf16* bs = bSrc + k0;
        __bf16* ad = &Alds[buf][dOff];
        __bf16* bd = &Blds[buf][dOff];
#if USE_ASYNC
        __builtin_amdgcn_global_load_async_to_lds_b128(to_gbl(as),     to_lds(ad),     0, 0);
        __builtin_amdgcn_global_load_async_to_lds_b128(to_gbl(as + 8), to_lds(ad + 8), 0, 0);
        __builtin_amdgcn_global_load_async_to_lds_b128(to_gbl(bs),     to_lds(bd),     0, 0);
        __builtin_amdgcn_global_load_async_to_lds_b128(to_gbl(bs + 8), to_lds(bd + 8), 0, 0);
#else
        v8bf a0 = *(const v8bf*)as, a1 = *(const v8bf*)(as + 8);
        v8bf b0 = *(const v8bf*)bs, b1 = *(const v8bf*)(bs + 8);
        *(v8bf*)ad       = a0; *(v8bf*)(ad + 8) = a1;
        *(v8bf*)bd       = b0; *(v8bf*)(bd + 8) = b1;
#endif
    };

    auto compute = [&](int buf) {
        v16bf a[4], b[2];
#pragma unroll
        for (int i = 0; i < 4; ++i) {
            const int r = mw * 64 + i * 16 + ln;
            v8bf alo = *(const v8bf*)&Alds[buf][r * 32 + g * 16];
            v8bf ahi = *(const v8bf*)&Alds[buf][r * 32 + g * 16 + 8];
#pragma unroll
            for (int e = 0; e < 8; ++e) { a[i][e] = alo[e]; a[i][e + 8] = ahi[e]; }
        }
#pragma unroll
        for (int j = 0; j < 2; ++j) {
            const int n = nw * 32 + j * 16 + ln;
            b[j] = *(const v16bf*)&Blds[buf][n * 32 + g * 16];
        }
#pragma unroll
        for (int i = 0; i < 4; ++i)
#pragma unroll
            for (int j = 0; j < 2; ++j)
                acc[i][j] = __builtin_amdgcn_wmma_f32_16x16x32_bf16(
                    false, a[i], false, b[j], (short)0, acc[i][j], false, false);
    };

    stage(0, 0);
#if USE_ASYNC
    __builtin_amdgcn_s_wait_asynccnt(0);
#endif
    __syncthreads();

    int buf = 0;
    for (int k0 = 0; k0 < Kdim; k0 += 32) {
        const int kn = k0 + 32;
        if (kn < Kdim) stage(buf ^ 1, kn);
        compute(buf);
#if USE_ASYNC
        __builtin_amdgcn_s_wait_asynccnt(0);
#endif
        __syncthreads();
        buf ^= 1;
    }

    // ---- epilogue: bias -> gelu -> residual -> store ----
#pragma unroll
    for (int i = 0; i < 4; ++i) {
#pragma unroll
        for (int j = 0; j < 2; ++j) {
#pragma unroll
            for (int r = 0; r < 8; ++r) {
                const int m = mBase + mw * 64 + i * 16 + g * 8 + r;
                const int n = nBase + nw * 32 + j * 16 + ln;
                float v = acc[i][j][r];
                if (bias)     v += bias[n];
                if (doGelu)   v = 0.5f * v * (1.0f + erff(v * 0.70710678118f));
                if (residual) v += residual[(size_t)m * Ndim + n];
                const size_t o = (size_t)m * Ndim + n;
                if (outF) outF[o] = v;
                if (outB) outB[o] = (__bf16)v;
            }
        }
    }
}

// =====================================================================
// 5) WMMA flash attention. One wave per (b, h, 16-query tile).
//    Per 32-key block: S = Q K^T (4 WMMAs, f32 acc), online softmax in
//    f32 on C-layout registers, P transposed via wave-private LDS to
//    the A-operand layout, O += P V (4 WMMAs) with running rescale.
//    qb/kb: bf16 [B*T, C] ([H][HD] inner); vt: bf16 [B*H][HD][T].
// =====================================================================
__global__ __launch_bounds__(256) void attention_wmma_kernel(
    const __bf16* __restrict__ qb, const __bf16* __restrict__ kb,
    const __bf16* __restrict__ vt, __bf16* __restrict__ ob)
{
    __shared__ __bf16 Plds[8][16 * 32];        // 1KB per wave
    const int tid  = threadIdx.x, lane = tid & 31, wv = tid >> 5;
    const int gw   = blockIdx.x * 8 + wv;      // 0 .. B*H*(T/16)-1
    const int qt   = gw & 63;                  // query tile (T/16 = 64)
    const int bh   = gw >> 6;                  // b*NH + h
    const int b    = bh >> 4, h = bh & 15;
    const int qt0  = qt * 16;
    const int g    = lane >> 4, ln = lane & 15;
    const float scale = 0.125f;                // HD^-0.5

    const size_t rowBase = (size_t)b * TSEQ * CDIM + (size_t)h * HDIM;

    // ---- Q A-fragments (rows m=ln), HD=64 -> two 16x32 fragments ----
    v16bf qa[2];
    {
        const __bf16* qrow = qb + rowBase + (size_t)(qt0 + ln) * CDIM;
#pragma unroll
        for (int i = 0; i < 2; ++i) {
            v8bf lo = *(const v8bf*)(qrow + i * 32 + g * 16);
            v8bf hi = *(const v8bf*)(qrow + i * 32 + g * 16 + 8);
#pragma unroll
            for (int e = 0; e < 8; ++e) { qa[i][e] = lo[e]; qa[i][e + 8] = hi[e]; }
        }
    }
    const __bf16* vhead = vt + (size_t)bh * HDIM * TSEQ;   // [d][t]

    v8f o[4] = {};                 // O tiles over d = {0..15,16..31,32..47,48..63}
    float mrow[8], lrow[8];
#pragma unroll
    for (int r = 0; r < 8; ++r) { mrow[r] = -INFINITY; lrow[r] = 0.f; }

    const int nblk = (qt0 + 16 + 31) >> 5;     // 32-key blocks (causal)
    for (int j0 = 0; j0 < nblk * 32; j0 += 32) {
        // ---- S = Q K^T : two 16x16 key tiles ----
        v8f s0 = {}, s1 = {};
#pragma unroll
        for (int i = 0; i < 2; ++i) {          // HD halves: d in [32i, 32i+32)
            v16bf kf0 = *(const v16bf*)(kb + rowBase + (size_t)(j0 + ln)      * CDIM + i * 32 + g * 16);
            v16bf kf1 = *(const v16bf*)(kb + rowBase + (size_t)(j0 + 16 + ln) * CDIM + i * 32 + g * 16);
            s0 = __builtin_amdgcn_wmma_f32_16x16x32_bf16(false, qa[i], false, kf0, (short)0, s0, false, false);
            s1 = __builtin_amdgcn_wmma_f32_16x16x32_bf16(false, qa[i], false, kf1, (short)0, s1, false, false);
        }
        // ---- causal mask + online softmax (row = qt0 + g*8 + r) ----
#pragma unroll
        for (int r = 0; r < 8; ++r) {
            const int qrow = qt0 + g * 8 + r;
            float v0 = s0[r] * scale;
            float v1 = s1[r] * scale;
            if (j0 + ln > qrow)      v0 = -INFINITY;
            if (j0 + 16 + ln > qrow) v1 = -INFINITY;
            float rmax = fmaxf(v0, v1);
#pragma unroll
            for (int msk = 8; msk > 0; msk >>= 1) rmax = fmaxf(rmax, __shfl_xor(rmax, msk, 32));
            const float mn   = fmaxf(mrow[r], rmax);
            const float corr = __expf(mrow[r] - mn);
            const float p0 = __expf(v0 - mn), p1 = __expf(v1 - mn);
            float rsum = p0 + p1;
#pragma unroll
            for (int msk = 8; msk > 0; msk >>= 1) rsum += __shfl_xor(rsum, msk, 32);
            lrow[r] = lrow[r] * corr + rsum;
            mrow[r] = mn;
            o[0][r] *= corr; o[1][r] *= corr; o[2][r] *= corr; o[3][r] *= corr;
            Plds[wv][(g * 8 + r) * 32 + ln]      = (__bf16)p0;
            Plds[wv][(g * 8 + r) * 32 + 16 + ln] = (__bf16)p1;
        }
        // ---- P as A-fragment (rows m=ln, kdim = 32 keys) via LDS ----
        v16bf pa;
        {
            v8bf lo = *(const v8bf*)&Plds[wv][ln * 32 + g * 16];
            v8bf hi = *(const v8bf*)&Plds[wv][ln * 32 + g * 16 + 8];
#pragma unroll
            for (int e = 0; e < 8; ++e) { pa[e] = lo[e]; pa[e + 8] = hi[e]; }
        }
        // ---- O += P V : B-frag b[i] = V[key=j0+16g+i][d=s*16+ln] ----
#pragma unroll
        for (int s = 0; s < 4; ++s) {
            v16bf vf = *(const v16bf*)(vhead + (size_t)(s * 16 + ln) * TSEQ + j0 + 16 * g);
            o[s] = __builtin_amdgcn_wmma_f32_16x16x32_bf16(false, pa, false, vf, (short)0, o[s], false, false);
        }
    }

    // ---- normalize + write (row = qt0 + g*8 + r, col d = s*16 + ln) ----
#pragma unroll
    for (int r = 0; r < 8; ++r) {
        const float invl = 1.0f / lrow[r];
        __bf16* orow = ob + rowBase + (size_t)(qt0 + g * 8 + r) * CDIM;
#pragma unroll
        for (int s = 0; s < 4; ++s)
            orow[s * 16 + ln] = (__bf16)(o[s][r] * invl);
    }
}

// =====================================================================
// host-side orchestration
// =====================================================================
static void launch_cvtT(const float* W, __bf16* Wt, int K, int N, hipStream_t s) {
    dim3 grid(N / 32, K / 32);
    cvt_transpose_kernel<<<grid, 256, 0, s>>>(W, Wt, K, N);
}
static void launch_gemm(const __bf16* A, const __bf16* Wt, const float* bias,
                        const float* res, float* oF, __bf16* oB,
                        int M, int N, int K, int gelu, hipStream_t s) {
    dim3 grid(N / 128, M / 128);
    gemm_bf16_wmma<<<grid, 256, 0, s>>>(A, Wt, bias, res, oF, oB, M, N, K, gelu);
}

extern "C" void kernel_launch(void* const* d_in, const int* in_sizes, int n_in,
                              void* d_out, int out_size, void* d_ws, size_t ws_size,
                              hipStream_t stream) {
    const int*   idx  = (const int*)  d_in[0];
    const float* emb  = (const float*)d_in[1];
    const float* pos  = (const float*)d_in[2];
    const float* Wq   = (const float*)d_in[3];
    const float* Wk   = (const float*)d_in[4];
    const float* Wv   = (const float*)d_in[5];
    const float* Wo   = (const float*)d_in[6];
    const float* bo   = (const float*)d_in[7];
    const float* W1   = (const float*)d_in[8];
    const float* b1   = (const float*)d_in[9];
    const float* W2   = (const float*)d_in[10];
    const float* b2   = (const float*)d_in[11];
    const float* ln1g = (const float*)d_in[12];
    const float* ln1b = (const float*)d_in[13];
    const float* ln2g = (const float*)d_in[14];
    const float* ln2b = (const float*)d_in[15];
    const float* lnfg = (const float*)d_in[16];
    const float* lnfb = (const float*)d_in[17];
    const float* Wlm  = (const float*)d_in[18];
    const float* blm  = (const float*)d_in[19];
    float* out = (float*)d_out;

    // ---- workspace carve-up ----
    char* ws = (char*)d_ws;
    float*  xb = (float*)ws;  ws += (size_t)MROWS * CDIM * 4;   // residual stream
    __bf16* hb = (__bf16*)ws; ws += (size_t)MROWS * CDIM * 2;   // LN output (A)
    __bf16* qbuf = (__bf16*)ws; ws += (size_t)MROWS * CDIM * 2; // Q bf16
    __bf16* kbuf = (__bf16*)ws; ws += (size_t)MROWS * CDIM * 2; // K bf16
    __bf16* vbuf = (__bf16*)ws; ws += (size_t)MROWS * CDIM * 2; // V bf16
    __bf16* vtb  = (__bf16*)ws; ws += (size_t)MROWS * CDIM * 2; // V^T per head
    __bf16* obuf = (__bf16*)ws; ws += (size_t)MROWS * CDIM * 2; // attn out (A)
    __bf16* gb = (__bf16*)ws; ws += (size_t)MROWS * FFD  * 2;   // gelu out (A)
    __bf16* wb = (__bf16*)ws; ws += (size_t)CDIM  * VSZ  * 2;   // bf16 W^T buf

    embed_kernel<<<MROWS, 256, 0, stream>>>(idx, emb, pos, xb);

    for (int l = 0; l < NL; ++l) {
        const size_t wcc = (size_t)l * CDIM * CDIM;
        const size_t wcf = (size_t)l * CDIM * FFD;
        // LN1 -> QKV projections (bf16 outputs)
        ln_bf16_kernel<<<MROWS, 256, 0, stream>>>(xb, ln1g + l * CDIM, ln1b + l * CDIM, hb);
        launch_cvtT(Wq + wcc, wb, CDIM, CDIM, stream);
        launch_gemm(hb, wb, nullptr, nullptr, nullptr, qbuf, MROWS, CDIM, CDIM, 0, stream);
        launch_cvtT(Wk + wcc, wb, CDIM, CDIM, stream);
        launch_gemm(hb, wb, nullptr, nullptr, nullptr, kbuf, MROWS, CDIM, CDIM, 0, stream);
        launch_cvtT(Wv + wcc, wb, CDIM, CDIM, stream);
        launch_gemm(hb, wb, nullptr, nullptr, nullptr, vbuf, MROWS, CDIM, CDIM, 0, stream);
        // per-head V transpose, then WMMA flash attention
        {
            dim3 vg(TSEQ / 32, NB * NH * (HDIM / 32));
            vtrans_kernel<<<vg, 256, 0, stream>>>(vbuf, vtb);
        }
        attention_wmma_kernel<<<(NB * NH * (TSEQ / 16)) / 8, 256, 0, stream>>>(qbuf, kbuf, vtb, obuf);
        // output projection + residual
        launch_cvtT(Wo + wcc, wb, CDIM, CDIM, stream);
        launch_gemm(obuf, wb, bo + l * CDIM, xb, xb, nullptr, MROWS, CDIM, CDIM, 0, stream);
        // LN2 -> MLP
        ln_bf16_kernel<<<MROWS, 256, 0, stream>>>(xb, ln2g + l * CDIM, ln2b + l * CDIM, hb);
        launch_cvtT(W1 + wcf, wb, CDIM, FFD, stream);
        launch_gemm(hb, wb, b1 + l * FFD, nullptr, nullptr, gb, MROWS, FFD, CDIM, 1, stream);
        launch_cvtT(W2 + wcf, wb, FFD, CDIM, stream);
        launch_gemm(gb, wb, b2 + l * CDIM, xb, xb, nullptr, MROWS, CDIM, FFD, 0, stream);
    }

    // final LN + LM head (N = 32000 = 250 * 128)
    ln_bf16_kernel<<<MROWS, 256, 0, stream>>>(xb, lnfg, lnfb, hb);
    launch_cvtT(Wlm, wb, CDIM, VSZ, stream);
    launch_gemm(hb, wb, blm, nullptr, out, nullptr, MROWS, VSZ, CDIM, 0, stream);
}
